// AuxiliaryGovernedAttention_23390391894556
// MI455X (gfx1250) — matrix operand order repaired and checked by reading, not compiled, not run on hardware
//
#include <hip/hip_runtime.h>
#include <stdint.h>

#define H   4096
#define S   2048
#define BB  2
#define NHEADS 32
#define NSLOT  100
#define DD  64
#define ROWS (BB*S)   // 4096 flattened (b,s) rows
#define KPAD 128      // slots padded to WMMA-friendly K

typedef __attribute__((ext_vector_type(16))) __bf16 v16bf;
typedef __attribute__((ext_vector_type(8)))  __bf16 v8bf;
typedef __attribute__((ext_vector_type(8)))  float  v8f;
typedef __attribute__((ext_vector_type(4)))  float  v4f;

static __device__ __forceinline__ unsigned short f2bf_bits(float f) {
  union { float f; uint32_t u; } v; v.f = f;
  uint32_t u = v.u;
  uint32_t r = u + 0x7FFFu + ((u >> 16) & 1u);   // round-to-nearest-even
  return (unsigned short)(r >> 16);
}
static __device__ __forceinline__ __bf16 bf_from_bits(unsigned short b) {
  __bf16 r; __builtin_memcpy(&r, &b, 2); return r;
}
static __device__ __forceinline__ __bf16 f2bf(float f) {
  return bf_from_bits(f2bf_bits(f));
}
// Packed pair conversion: prefer v_cvt_pk_bf16_f32 when the builtin exists.
static __device__ __forceinline__ void cvt2bf(v16bf& a, int i, float x, float y) {
#if __has_builtin(__builtin_amdgcn_cvt_pk_bf16_f32)
  auto pk = __builtin_amdgcn_cvt_pk_bf16_f32(x, y);   // 2 x bf16 in 32 bits
  unsigned int bits; __builtin_memcpy(&bits, &pk, 4);
  a[i]     = bf_from_bits((unsigned short)(bits & 0xFFFFu));
  a[i + 1] = bf_from_bits((unsigned short)(bits >> 16));
#else
  a[i] = f2bf(x); a[i + 1] = f2bf(y);
#endif
}
static __device__ __forceinline__ v8f wmma_bf16(v16bf a, v16bf b, v8f c) {
  return __builtin_amdgcn_wmma_f32_16x16x32_bf16(false, a, false, b, (short)0, c, false, false);
}

// ---------------- Kernel 1: entropy of head-averaged attention + gate -------
// One block per (b,s) row; streams 32 heads x 2048 cols = 256KB with NT b128.
__global__ void __launch_bounds__(256)
entropy_gate_kernel(const float* __restrict__ w, const float* __restrict__ gw1,
                    const float* __restrict__ gb, float* __restrict__ gate) {
  int row = blockIdx.x;                 // 0..ROWS-1
  int b = row / S, s = row % S;
  long base0 = ((long)(b * NHEADS) * S + s) * (long)S;
  int t = threadIdx.x;                  // 256 threads, 2 x float4 each
  v4f a0 = {}, a1 = {};
  for (int h = 0; h < NHEADS; ++h) {
    const float* p = w + base0 + (long)h * (long)S * (long)S;
    a0 += __builtin_nontemporal_load((const v4f*)p + t);
    a1 += __builtin_nontemporal_load((const v4f*)(p + 1024) + t);
  }
  float ent = 0.f;
#pragma unroll
  for (int i = 0; i < 4; ++i) {
    float avg = a0[i] * (1.0f / NHEADS);
    ent -= avg * __logf(avg + 1e-10f);
    avg = a1[i] * (1.0f / NHEADS);
    ent -= avg * __logf(avg + 1e-10f);
  }
  __shared__ float red[256];
  red[t] = ent; __syncthreads();
  for (int off = 128; off > 0; off >>= 1) {
    if (t < off) red[t] += red[t + off];
    __syncthreads();
  }
  if (t == 0) {
    float e = red[0];
    float g = 1.0f / (1.0f + __expf(-(gw1[0] * e + gb[0])));
    if (e < 0.5f) g = 0.f;                       // TAU_LOW veto
    else if (e > 2.0f) g = fminf(g, 0.8f);       // TAU_HIGH cap
    gate[row] = g;
  }
}

// ---------------- Kernel 2/3: bf16 operand prep --------------------------
__global__ void prep_wq(const float* __restrict__ wq, unsigned short* __restrict__ wqb) {
  int i = blockIdx.x * blockDim.x + threadIdx.x;   // D*H
  wqb[i] = f2bf_bits(wq[i]);
}
// aux_values [NS,H] fp32 -> avT [H][KPAD] bf16 (transposed, zero-padded K)
__global__ void prep_avT(const float* __restrict__ av, unsigned short* __restrict__ avT) {
  int i = blockIdx.x * blockDim.x + threadIdx.x;   // H*KPAD
  int h = i >> 7, n = i & (KPAD - 1);
  avT[i] = (n < NSLOT) ? f2bf_bits(av[(size_t)n * H + h]) : (unsigned short)0;
}

// ---------------- Kernel 4: query WMMA + scores + softmax -> attn(bf16) ---
// Block = 4 waves, 16 output rows. Wave w computes query cols [16w,16w+16)
// over K=4096 via v_wmma_f32_16x16x32_bf16, then block does scores+softmax.
__global__ void __launch_bounds__(128)
qscore_kernel(const float* __restrict__ hidden, const unsigned short* __restrict__ wqb,
              const float* __restrict__ akeys, const float* __restrict__ rel,
              unsigned short* __restrict__ attnb) {
  __shared__ float ldsq[16 * 64];
  __shared__ float ldss[16 * 104];
  int m0 = blockIdx.x * 16;
  int t = threadIdx.x, lane = t & 31, wv = t >> 5;
  int half = lane >> 4, ncol = lane & 15;

  const float* ah = hidden + (size_t)(m0 + ncol) * H;            // A row (M = lane&15)
  const unsigned short* bp = wqb + (size_t)(wv * 16 + ncol) * H; // B col (N = lane&15)
  v8f c = {};
  for (int k0 = 0; k0 < H; k0 += 32) {
    int o1 = k0 + half * 16;          // A lane chunk: [o1..o1+15] contiguous K
    v4f f0 = *(const v4f*)(ah + o1);
    v4f f1 = *(const v4f*)(ah + o1 + 4);
    v4f f2 = *(const v4f*)(ah + o1 + 8);
    v4f f3 = *(const v4f*)(ah + o1 + 12);
    v16bf a;
#pragma unroll
    for (int i = 0; i < 4; i += 2) {
      cvt2bf(a, i,      f0[i], f0[i + 1]);
      cvt2bf(a, 4 + i,  f1[i], f1[i + 1]);
      cvt2bf(a, 8 + i,  f2[i], f2[i + 1]);
      cvt2bf(a, 12 + i, f3[i], f3[i + 1]);
    }
    v16bf bfr = *(const v16bf*)(const void*)(bp + k0 + half * 16); // 16 contiguous K
    c = wmma_bf16(a, bfr, c);
  }
  // spill query tile [16,64] to LDS (C layout: VGPR r -> M=r+8*half, N=lane&15)
#pragma unroll
  for (int r = 0; r < 8; ++r)
    ldsq[(r + 8 * half) * 64 + wv * 16 + ncol] = c[r];
  __syncthreads();

  // scores[m][n] = q.keys/8 + log(rel+1e-10)
  {
    int m = t & 15;
    for (int n = t >> 4; n < NSLOT; n += 8) {
      float s = 0.f;
#pragma unroll 8
      for (int d = 0; d < DD; ++d) s += ldsq[m * 64 + d] * akeys[n * DD + d];
      ldss[m * 104 + n] = s * 0.125f + __logf(rel[n] + 1e-10f);
    }
  }
  __syncthreads();
  if (t < 16) {
    int m = t;
    float mx = -3.4e38f;
    for (int n = 0; n < NSLOT; ++n) mx = fmaxf(mx, ldss[m * 104 + n]);
    float sum = 0.f;
    for (int n = 0; n < NSLOT; ++n) sum += __expf(ldss[m * 104 + n] - mx);
    float inv = 1.0f / sum;
    unsigned short* ap = attnb + (size_t)(m0 + m) * KPAD;
    for (int n = 0; n < NSLOT; ++n)
      ap[n] = f2bf_bits(__expf(ldss[m * 104 + n] - mx) * inv);
    for (int n = NSLOT; n < KPAD; ++n) ap[n] = 0;   // zero K padding
  }
}

// ---------------- Kernel 5: aux output WMMA + gated fusion ----------------
// Block = 8 waves, 16 rows x 128 cols; each wave does a 16x16 C tile, K=128.
__global__ void __launch_bounds__(256)
fuse_kernel(const float* __restrict__ primary, const unsigned short* __restrict__ attnb,
            const unsigned short* __restrict__ avT, const float* __restrict__ gate,
            float* __restrict__ out) {
  __shared__ float ldsg[16];
  int m0 = blockIdx.x * 16;
  int t = threadIdx.x;
  if (t < 16) ldsg[t] = gate[m0 + t];
  __syncthreads();
  int lane = t & 31, wv = t >> 5;
  int half = lane >> 4, ncol = lane & 15;
  int h0 = blockIdx.y * 128 + wv * 16;

  const unsigned short* ap = attnb + (size_t)(m0 + ncol) * KPAD;  // A row
  const unsigned short* bp = avT + (size_t)(h0 + ncol) * KPAD;    // B col
  v8f c = {};
#pragma unroll
  for (int k0 = 0; k0 < KPAD; k0 += 32) {
    int o1 = k0 + half * 16;
    v8bf alo = *(const v8bf*)(const void*)(ap + o1);
    v8bf ahi = *(const v8bf*)(const void*)(ap + o1 + 8);
    v16bf a;
#pragma unroll
    for (int i = 0; i < 8; ++i) { a[i] = alo[i]; a[8 + i] = ahi[i]; }
    v16bf bfr = *(const v16bf*)(const void*)(bp + k0 + half * 16);
    c = wmma_bf16(a, bfr, c);
  }
#pragma unroll
  for (int r = 0; r < 8; ++r) {
    int m = r + 8 * half;
    size_t idx = (size_t)(m0 + m) * H + h0 + ncol;
    out[idx] = primary[idx] + ldsg[m] * c[r];
  }
}

extern "C" void kernel_launch(void* const* d_in, const int* in_sizes, int n_in,
                              void* d_out, int out_size, void* d_ws, size_t ws_size,
                              hipStream_t stream) {
  const float* hidden  = (const float*)d_in[0];
  const float* primary = (const float*)d_in[1];
  const float* paw     = (const float*)d_in[2];
  const float* rel     = (const float*)d_in[3];
  const float* wq      = (const float*)d_in[4];
  const float* akeys   = (const float*)d_in[5];
  const float* avals   = (const float*)d_in[6];
  const float* gw1     = (const float*)d_in[7];
  const float* gb      = (const float*)d_in[8];
  float* out = (float*)d_out;

  char* ws = (char*)d_ws;
  float* gate          = (float*)ws;                                   // 16 KB
  unsigned short* wqb  = (unsigned short*)(ws + 16384);                // 512 KB
  unsigned short* avT  = (unsigned short*)(ws + 16384 + 524288);       // 1 MB
  unsigned short* attb = (unsigned short*)(ws + 16384 + 524288 + 1048576); // 1 MB

  entropy_gate_kernel<<<dim3(ROWS), dim3(256), 0, stream>>>(paw, gw1, gb, gate);
  prep_wq<<<dim3((DD * H) / 256), dim3(256), 0, stream>>>(wq, wqb);
  prep_avT<<<dim3((H * KPAD) / 256), dim3(256), 0, stream>>>(avals, avT);
  qscore_kernel<<<dim3(ROWS / 16), dim3(128), 0, stream>>>(hidden, wqb, akeys, rel, attb);
  fuse_kernel<<<dim3(ROWS / 16, H / 128), dim3(256), 0, stream>>>(primary, attb, avT, gate, out);
}
